// CTBG_25434796327649
// MI455X (gfx1250) — matrix operand order repaired and checked by reading, not compiled, not run on hardware
//
#include <hip/hip_runtime.h>

typedef __bf16 bf16;
typedef __attribute__((ext_vector_type(16))) __bf16 v16bf;
typedef __attribute__((ext_vector_type(8)))  __bf16 bf16x8;
typedef __attribute__((ext_vector_type(4)))  __bf16 bf16x4;
typedef __attribute__((ext_vector_type(8)))  float  v8f;
typedef __attribute__((ext_vector_type(4)))  float  f32x4;

#define B_ROWS 16384
#define D1     1536      // GPe in/out dim, GPi out dim
#define D2     3072      // GPi in dim
#define U3     512
#define ACT    6

#define BM 128
#define BN 128
#define BK 32
#define LDSK (BK + 8)    // 40 halves = 80B row stride -> 20-bank stride, conflict-free
#define NBUF 3           // 3-deep LDS pipeline, DMA prefetch depth 2
#define BUFELEMS (BM * LDSK)

union Frag { v16bf v; bf16x8 h[2]; };

// ---------------------------------------------------------------- prep kernels

__global__ __launch_bounds__(256)
void cvt_f32_to_bf16(const float* __restrict__ in, bf16* __restrict__ out, int n4) {
    int i = blockIdx.x * blockDim.x + threadIdx.x;
    if (i >= n4) return;
    f32x4 v = ((const f32x4*)in)[i];
    bf16x4 o;
    o[0] = (bf16)v[0]; o[1] = (bf16)v[1]; o[2] = (bf16)v[2]; o[3] = (bf16)v[3];
    ((bf16x4*)out)[i] = o;
}

// w: [K][N] fp32 row-major; mask: [N][K] fp32 or nullptr; out: [N][K] bf16
__global__ __launch_bounds__(256)
void mask_transpose_bf16(const float* __restrict__ w, const float* __restrict__ mask,
                         bf16* __restrict__ out, int K, int N) {
    long total = (long)K * N;
    long i = (long)blockIdx.x * blockDim.x + threadIdx.x;
    if (i >= total) return;
    int k = (int)(i % K);
    int n = (int)(i / K);
    float v = w[(long)k * N + n];
    if (mask) v *= mask[i];          // mask is [N][K] row-major; i == n*K + k
    out[i] = (bf16)v;
}

// ---------------------------------------------------------------- WMMA GEMM
// C[M,N] = act( [A0 | A1][M, K0+K1] @ WT[N, K0+K1]^T + bias )
// A0/A1/WT bf16, fp32 accumulate via v_wmma_f32_16x16x32_bf16.
// Tiles staged memory->LDS with CDNA5 async DMA (global_load_async_to_lds_b128,
// ASYNCcnt-tracked, in-order retirement), 3 LDS buffers addressed as
// base + rotating element offset (no per-buffer register arrays), prefetch
// depth 2. Steady loop is branch-free: s_wait_asynccnt 4 (tile kt landed,
// kt+1 in flight) -> barrier -> DMA tile kt+2 -> 8x WMMA. Last 2 tiles peeled.
// Requires: M % 128 == 0, N % 128 == 0, K0 % 32 == 0, K1 % 32 == 0, K/32 >= 2.

template<int RELU>
__global__ __launch_bounds__(256)
void gemm_bf16_wmma(const bf16* __restrict__ A0, int K0,
                    const bf16* __restrict__ A1, int K1,
                    const bf16* __restrict__ WT,
                    const float* __restrict__ bias,
                    bf16* __restrict__ C, int N)
{
    const int K = K0 + K1;
    __shared__ bf16 As[NBUF * BUFELEMS];
    __shared__ bf16 Bs[NBUF * BUFELEMS];

    const int tid  = threadIdx.x;
    const int lane = tid & 31;
    const int wave = tid >> 5;
    const int wm   = wave & 3;                 // 4 waves along M (32 rows each)
    const int wn   = wave >> 2;                // 2 waves along N (64 cols each)
    const long block_m = (long)blockIdx.y * BM;
    const long block_n = (long)blockIdx.x * BN;

    v8f acc[2][4] = {};

    const int nkt = K / BK;

    // ---- per-thread staging addresses, buffer 0 (2 chunks of 8 halves each)
    const bf16* pA0[2];
    const bf16* pA1[2];
    const bf16* pW[2];
    unsigned lds_a0[2], lds_b0[2];             // byte addresses within buffer 0
    #pragma unroll
    for (int c0 = 0; c0 < 2; ++c0) {
        int c   = tid + c0 * 256;              // 512 chunks cover a 128x32 tile
        int row = c >> 2;                      // 0..127
        int k8  = (c & 3) * 8;                 // 0,8,16,24
        pA0[c0] = A0 + (block_m + row) * (long)K0 + k8;
        pA1[c0] = A1 + (block_m + row) * (long)K1 + k8;
        pW[c0]  = WT + (block_n + row) * (long)K + k8;
        lds_a0[c0] = (unsigned)(size_t)(const void*)&As[row * LDSK + k8];
        lds_b0[c0] = (unsigned)(size_t)(const void*)&Bs[row * LDSK + k8];
    }

    // ---- async DMA stage into buffer at element offset 'boff_el'
    auto stage_async = [&](unsigned boff_el, int kt) {
        const int kk = kt * BK;
        #pragma unroll
        for (int c0 = 0; c0 < 2; ++c0) {
            const bf16* ga = (kk < K0) ? (pA0[c0] + kk) : (pA1[c0] + (kk - K0));
            unsigned la = lds_a0[c0] + boff_el * 2u;
            asm volatile("global_load_async_to_lds_b128 %0, %1, off"
                         :: "v"(la), "v"(ga) : "memory");
            const bf16* gw = pW[c0] + kk;
            unsigned lb = lds_b0[c0] + boff_el * 2u;
            asm volatile("global_load_async_to_lds_b128 %0, %1, off"
                         :: "v"(lb), "v"(gw) : "memory");
        }
    };

    const int half = lane >> 4;                // 0 or 1
    const int l16  = lane & 15;

    // ---- fragment element indices, buffer 0
    unsigned aidx[2], bidx[4];
    #pragma unroll
    for (int s = 0; s < 2; ++s) aidx[s] = (wm * 32 + s * 16 + l16) * LDSK + half * 8;
    #pragma unroll
    for (int t = 0; t < 4; ++t) bidx[t] = (wn * 64 + t * 16 + l16) * LDSK + half * 8;

    // ---- one k-step: 2 A frags + 4 B frags from LDS, 8 WMMAs
    auto compute_tile = [&](unsigned cur_el) {
        // A fragments: 16x32 bf16. lanes 0-15 hold K[0..7] (v0-3), K[16..23] (v4-7);
        // lanes 16-31 hold K[8..15], K[24..31].  -> two contiguous b128 LDS reads.
        Frag af[2];
        #pragma unroll
        for (int s = 0; s < 2; ++s) {
            const bf16* p = &As[aidx[s] + cur_el];
            af[s].h[0] = *(const bf16x8*)(p);
            af[s].h[1] = *(const bf16x8*)(p + 16);
        }
        // B fragments: 32x16 bf16, staged transposed ([n][k]) in LDS.
        // lanes 0-15: K[0..7], K[16..23] of column n; lanes 16-31: K[8..15], K[24..31].
        Frag bfr[4];
        #pragma unroll
        for (int t = 0; t < 4; ++t) {
            const bf16* p = &Bs[bidx[t] + cur_el];
            bfr[t].h[0] = *(const bf16x8*)(p);
            bfr[t].h[1] = *(const bf16x8*)(p + 16);
        }
        #pragma unroll
        for (int s = 0; s < 2; ++s)
            #pragma unroll
            for (int t = 0; t < 4; ++t)
                acc[s][t] = __builtin_amdgcn_wmma_f32_16x16x32_bf16(
                    false, af[s].v, false, bfr[t].v,
                    (short)0, acc[s][t], false, false);
    };

    stage_async(0, 0);
    stage_async(BUFELEMS, 1);

    unsigned cur_el = 0;                       // buffer offset of tile kt
    unsigned pre_el = 2 * BUFELEMS;            // buffer offset for tile kt+2

    // ---- steady state: branch-free
    for (int kt = 0; kt < nkt - 2; ++kt) {
        asm volatile("s_wait_asynccnt 4" ::: "memory");   // tile kt landed
        __syncthreads();                       // publish LDS; all reads of 'pre' done
        stage_async(pre_el, kt + 2);
        compute_tile(cur_el);
        cur_el = (cur_el == (NBUF - 1) * BUFELEMS) ? 0u : cur_el + BUFELEMS;
        pre_el = (pre_el == (NBUF - 1) * BUFELEMS) ? 0u : pre_el + BUFELEMS;
    }
    // ---- drain: tile nkt-2 (tile nkt-1 still in flight), then tile nkt-1
    asm volatile("s_wait_asynccnt 4" ::: "memory");
    __syncthreads();
    compute_tile(cur_el);
    cur_el = (cur_el == (NBUF - 1) * BUFELEMS) ? 0u : cur_el + BUFELEMS;

    asm volatile("s_wait_asynccnt 0" ::: "memory");
    __syncthreads();
    compute_tile(cur_el);

    // Epilogue: C/D layout -> VGPR r, lanes 0-15: (m0+r, n0+lane); lanes 16-31: (m0+8+r, ...)
    #pragma unroll
    for (int t = 0; t < 4; ++t) {
        const long n_idx = block_n + wn * 64 + t * 16 + l16;
        const float bv = bias[n_idx];
        #pragma unroll
        for (int s = 0; s < 2; ++s) {
            const long m_base = block_m + wm * 32 + s * 16 + half * 8;
            #pragma unroll
            for (int r = 0; r < 8; ++r) {
                float v = acc[s][t][r] + bv;
                if (RELU) v = fmaxf(v, 0.0f);
                C[(m_base + r) * (long)N + n_idx] = (bf16)v;
            }
        }
    }
}

// ---------------------------------------------------------------- final layer
// out[M,6] = relu(H[M,512] @ W[512,6] + b). One wave per row, shuffle-reduce.

__global__ __launch_bounds__(256)
void head6(const bf16* __restrict__ H, const float* __restrict__ W,
           const float* __restrict__ b, float* __restrict__ out, int M)
{
    const int lane = threadIdx.x & 31;
    const int row  = blockIdx.x * 8 + (threadIdx.x >> 5);
    if (row >= M) return;
    const bf16* h = H + (long)row * U3;
    float acc[ACT] = {};
    for (int k = lane; k < U3; k += 32) {
        float hv = (float)h[k];
        const float* wr = W + k * ACT;
        #pragma unroll
        for (int j = 0; j < ACT; ++j) acc[j] += hv * wr[j];
    }
    #pragma unroll
    for (int off = 16; off > 0; off >>= 1)
        #pragma unroll
        for (int j = 0; j < ACT; ++j)
            acc[j] += __shfl_xor(acc[j], off, 32);
    if (lane < ACT)
        out[(long)row * ACT + lane] = fmaxf(acc[lane] + b[lane], 0.0f);
}

// ---------------------------------------------------------------- launch

extern "C" void kernel_launch(void* const* d_in, const int* in_sizes, int n_in,
                              void* d_out, int out_size, void* d_ws, size_t ws_size,
                              hipStream_t stream) {
    (void)in_sizes; (void)n_in; (void)out_size; (void)ws_size;

    const float* x        = (const float*)d_in[0];
    const float* gpe_mask = (const float*)d_in[1];
    const float* gpe_w    = (const float*)d_in[2];
    const float* gpe_b    = (const float*)d_in[3];
    const float* gpi_mask = (const float*)d_in[4];
    const float* gpi_w    = (const float*)d_in[5];
    const float* gpi_b    = (const float*)d_in[6];
    const float* w1       = (const float*)d_in[7];
    const float* b1       = (const float*)d_in[8];
    const float* w2       = (const float*)d_in[9];
    const float* b2       = (const float*)d_in[10];
    const float* w3       = (const float*)d_in[11];
    const float* b3       = (const float*)d_in[12];
    float* out            = (float*)d_out;

    char*  ws  = (char*)d_ws;
    size_t off = 0;
    auto alloc = [&](size_t bytes) -> void* {
        void* p = ws + off;
        off += (bytes + 255) & ~(size_t)255;
        return p;
    };
    bf16* xb   = (bf16*)alloc((size_t)B_ROWS * D1 * 2);
    bf16* mweT = (bf16*)alloc((size_t)D1 * D1 * 2);
    bf16* mwiT = (bf16*)alloc((size_t)D1 * D2 * 2);
    bf16* w1T  = (bf16*)alloc((size_t)U3 * D1 * 2);
    bf16* w2T  = (bf16*)alloc((size_t)U3 * U3 * 2);
    bf16* gpeo = (bf16*)alloc((size_t)B_ROWS * D1 * 2);
    bf16* gpio = (bf16*)alloc((size_t)B_ROWS * D1 * 2);
    bf16* h1   = (bf16*)alloc((size_t)B_ROWS * U3 * 2);
    bf16* h2   = (bf16*)alloc((size_t)B_ROWS * U3 * 2);

    // --- prep: x -> bf16; masked/plain weights -> transposed bf16 [N][K]
    {
        int n4 = (B_ROWS * D1) / 4;
        cvt_f32_to_bf16<<<(n4 + 255) / 256, 256, 0, stream>>>(x, xb, n4);
    }
    {
        long t = (long)D1 * D1;
        mask_transpose_bf16<<<(int)((t + 255) / 256), 256, 0, stream>>>(gpe_w, gpe_mask, mweT, D1, D1);
    }
    {
        long t = (long)D2 * D1;
        mask_transpose_bf16<<<(int)((t + 255) / 256), 256, 0, stream>>>(gpi_w, gpi_mask, mwiT, D2, D1);
    }
    {
        long t = (long)D1 * U3;
        mask_transpose_bf16<<<(int)((t + 255) / 256), 256, 0, stream>>>(w1, nullptr, w1T, D1, U3);
    }
    {
        long t = (long)U3 * U3;
        mask_transpose_bf16<<<(int)((t + 255) / 256), 256, 0, stream>>>(w2, nullptr, w2T, U3, U3);
    }

    // --- GEMM chain (concat in layer 2 folded as split-K over two A pointers)
    dim3 blk(256);
    gemm_bf16_wmma<0><<<dim3(D1 / BN, B_ROWS / BM), blk, 0, stream>>>(
        xb, D1, xb, 0, mweT, gpe_b, gpeo, D1);
    gemm_bf16_wmma<0><<<dim3(D1 / BN, B_ROWS / BM), blk, 0, stream>>>(
        xb, D1, gpeo, D1, mwiT, gpi_b, gpio, D1);
    gemm_bf16_wmma<1><<<dim3(U3 / BN, B_ROWS / BM), blk, 0, stream>>>(
        gpio, D1, gpio, 0, w1T, b1, h1, U3);
    gemm_bf16_wmma<1><<<dim3(U3 / BN, B_ROWS / BM), blk, 0, stream>>>(
        h1, U3, h1, 0, w2T, b2, h2, U3);

    // --- tiny final layer in fp32
    head6<<<(B_ROWS + 7) / 8, 256, 0, stream>>>(h2, w3, b3, out, B_ROWS);
}
